// Postprocessor_68161130988061
// MI455X (gfx1250) — matrix operand order repaired and checked by reading, not compiled, not run on hardware
//
#include <hip/hip_runtime.h>
#include <hip/hip_bf16.h>

// Problem constants (from reference): B=8, Q=150, C1=134 (C=133), H=W=256.
#define NB  8
#define NQ  150
#define NC1 134
#define NC  133
#define QP  160   // Q padded to 5 K-tiles of 32 for wmma_f32_16x16x32_bf16
#define MP  144   // C padded to 9 M-tiles of 16
#define HW  65536 // 256*256

typedef __attribute__((ext_vector_type(16))) __bf16 v16bf;
typedef __attribute__((ext_vector_type(8)))  __bf16 v8bf;
typedef __attribute__((ext_vector_type(8)))  float  v8f;

// ---------------------------------------------------------------------------
// Kernel 1: per-(b,q) softmax over 134 classes, store A^T in bf16 workspace
// laid out [b][c (0..143)][q (0..159)], zero-padded in both c and q so the
// GEMM kernel needs no K/M tail logic on A. One wave32 per (b,q) slot.
// Workspace use: 8*144*160*2 B = 368,640 bytes.
// ---------------------------------------------------------------------------
__global__ void softmax_to_A(const float* __restrict__ logits,
                             __bf16* __restrict__ A) {
    const int lane = threadIdx.x & 31;
    const int wid  = blockIdx.x * (blockDim.x >> 5) + (threadIdx.x >> 5); // 0..1279
    const int b = wid / QP;
    const int q = wid % QP;
    if (b >= NB) return;

    if (q >= NQ) { // padded K columns -> zeros
        #pragma unroll
        for (int i = 0; i < 5; ++i) {
            int c = lane + 32 * i;
            if (c < MP) A[((size_t)b * MP + c) * QP + q] = (__bf16)0.0f;
        }
        return;
    }

    const float* row = logits + ((size_t)b * NQ + q) * NC1;
    float v[5];
    float mx = -3.0e38f;
    #pragma unroll
    for (int i = 0; i < 5; ++i) {
        int c = lane + 32 * i;
        v[i] = (c < NC1) ? row[c] : -3.0e38f;
        mx = fmaxf(mx, v[i]);
    }
    #pragma unroll
    for (int off = 16; off > 0; off >>= 1) mx = fmaxf(mx, __shfl_xor(mx, off, 32));

    float s = 0.0f;
    #pragma unroll
    for (int i = 0; i < 5; ++i) {
        v[i] = __expf(v[i] - mx);      // exp(-huge) == 0 for padded lanes
        int c = lane + 32 * i;
        if (c < NC1) s += v[i]; else v[i] = 0.0f;
    }
    #pragma unroll
    for (int off = 16; off > 0; off >>= 1) s += __shfl_xor(s, off, 32);
    const float inv = __builtin_amdgcn_rcpf(s);  // single v_rcp_f32; ~1ulp is
                                                 // noise vs bf16 quantization

    #pragma unroll
    for (int i = 0; i < 5; ++i) {
        int c = lane + 32 * i;
        if (c < MP) {
            float p = (c < NC) ? v[i] * inv : 0.0f; // drop no-object class + M pad
            A[((size_t)b * MP + c) * QP + q] = (__bf16)p;
        }
    }
}

// ---------------------------------------------------------------------------
// Kernel 2: semseg[b, 0..132, n] = A[b] (144x160, bf16) @ sigmoid(masks[b]) (160x65536)
// Each wave32 owns one 16-column N slice and sweeps ALL 9 M-tiles, so every
// mask element is loaded from HBM + sigmoided exactly once (memory-bound:
// ~593 MB total HBM traffic -> ~25us at 23.3 TB/s). 9 f32 accumulators
// (72 VGPRs) + one shared B fragment per K-step; A fragments stream from the
// L2-resident 46KB/batch workspace. Sigmoid uses a single v_exp_f32 +
// v_rcp_f32 per element so VALU issue stays well under the 4/cycle/WGP budget
// at the ~0.45 loads/cycle/WGP rate HBM can sustain.
// ---------------------------------------------------------------------------
__global__ void __launch_bounds__(256)
semseg_wmma(const float* __restrict__ masks,
            const __bf16* __restrict__ A,
            float* __restrict__ out) {
    const int lane = threadIdx.x & 31;
    const int wave = threadIdx.x >> 5;
    const int b    = blockIdx.y;
    const int ncol = (blockIdx.x * 8 + wave) * 16 + (lane & 15);
    const int hi   = lane >> 4;          // 0 for lanes 0-15, 1 for lanes 16-31
    const int am   = lane & 15;          // A-matrix row owned by this lane
    const int ka   = hi ? 8 : 0;         // A K-offset per ISA 16-bit A layout
    const int kb   = hi ? 8 : 0;         // B K-offset matching A layout

    v8f acc[9];
    #pragma unroll
    for (int m = 0; m < 9; ++m)
        #pragma unroll
        for (int i = 0; i < 8; ++i) acc[m][i] = 0.0f;

    const float*  mb = masks + (size_t)b * NQ * HW;
    const __bf16* Ab = A + (size_t)b * MP * QP;

    for (int kt = 0; kt < 5; ++kt) {
        const int kbase = kt * 32;

        // B fragment (32x16 bf16): this lane holds K = {kb..kb+7, kb+16..kb+23} for column ncol.
        // Fused fast sigmoid; q >= 150 contributes 0 (A is also 0 there).
        v16bf bf;
        #pragma unroll
        for (int j = 0; j < 16; ++j) {
            const int q = kbase + kb + (j < 8 ? j : j + 8);
            float sgm = 0.0f;
            if (q < NQ) {
                float x = mb[(size_t)q * HW + ncol];
                sgm = __builtin_amdgcn_rcpf(1.0f + __expf(-x));
            }
            bf[j] = (__bf16)sgm;
        }

        // Accumulate all 9 M-tiles against this B fragment.
        #pragma unroll
        for (int mt = 0; mt < 9; ++mt) {
            const __bf16* arow = Ab + ((size_t)(mt * 16 + am)) * QP + kbase;
            // lanes 0-15: K = {0..7, 16..23}; lanes 16-31: K = {8..15, 24..31}
            v8bf alo = *(const v8bf*)(arow + ka);        // 16B-aligned b128 load
            v8bf ahi = *(const v8bf*)(arow + ka + 16);
            v16bf af;
            #pragma unroll
            for (int j = 0; j < 8; ++j) { af[j] = alo[j]; af[j + 8] = ahi[j]; }
            acc[mt] = __builtin_amdgcn_wmma_f32_16x16x32_bf16(
                false, af, false, bf, (short)0, acc[mt], false, false);
        }
    }

    // Store: C/D layout -> VGPR j holds row M = mt*16 + j + 8*hi, col = lane&15.
    float* ob = out + (size_t)b * NC * HW;
    #pragma unroll
    for (int mt = 0; mt < 9; ++mt) {
        #pragma unroll
        for (int j = 0; j < 8; ++j) {
            const int c = mt * 16 + j + 8 * hi;
            if (c < NC) ob[(size_t)c * HW + ncol] = acc[mt][j];
        }
    }
}

extern "C" void kernel_launch(void* const* d_in, const int* in_sizes, int n_in,
                              void* d_out, int out_size, void* d_ws, size_t ws_size,
                              hipStream_t stream) {
    const float* logits = (const float*)d_in[0]; // [8,150,134]
    const float* masks  = (const float*)d_in[1]; // [8,150,256,256]
    float*       out    = (float*)d_out;         // [8,133,256,256]
    __bf16*      Aws    = (__bf16*)d_ws;         // [8,144,160] bf16 (368,640 B)

    // 8*160 = 1280 (b,q) waves, 8 waves/block.
    softmax_to_A<<<160, 256, 0, stream>>>(logits, Aws);

    // grid.x: 65536 cols / (8 waves * 16 cols) = 512; grid.y: batch.
    semseg_wmma<<<dim3(512, 8), 256, 0, stream>>>(masks, Aws, out);
}
